// BasicBlock_8323646619714
// MI455X (gfx1250) — compile-verified, matrix operand order checked by reading, not run on hardware
//
#include <hip/hip_runtime.h>

#define B_ 2
#define N_ 4096
#define K_ 16
#define C_ 64
#define G_ 125
#define GC 8           // g-chunk: fp32 acc 16*8*64*4 = 32KB ; fp16 frags 24KB
#define EPS 1e-5f

typedef __attribute__((ext_vector_type(16))) _Float16 v16h;
typedef __attribute__((ext_vector_type(8)))  float    v8f;
typedef __attribute__((ext_vector_type(4)))  unsigned v4u;

union FragU { unsigned u[8]; v4u q[2]; v16h h; };

__device__ inline unsigned pk2h(float a, float b) {
  union { _Float16 h[2]; unsigned u; } x;
  x.h[0] = (_Float16)a; x.h[1] = (_Float16)b;
  return x.u;
}

// ---------------------------------------------------------------- alpha ----
// per (b,n,k): 8 nonzero trilinear cells (g index + weight)
__global__ void alpha_kernel(const float* __restrict__ xyz,
                             const int* __restrict__ knn,
                             int* __restrict__ aidx, float* __restrict__ aw) {
  int t = blockIdx.x * blockDim.x + threadIdx.x;
  if (t >= B_ * N_ * K_) return;
  int b = t / (N_ * K_);
  int r = t - b * (N_ * K_);
  int n = r / K_;
  int nb = knn[t];
  int j[3]; float w0[3], w1[3];
  #pragma unroll
  for (int d = 0; d < 3; ++d) {
    float xc = xyz[(b * 3 + d) * N_ + n];
    float xn = xyz[(b * 3 + d) * N_ + nb];
    float s = xn - xc;                       // RADIUS = 1
    s = fminf(1.f, fmaxf(-1.f, s));
    float u = (s + 1.f) * 2.f;               // in [0,4]
    int j0 = (int)floorf(u);
    j0 = j0 < 0 ? 0 : (j0 > 3 ? 3 : j0);
    float a1 = u - (float)j0;
    j[d] = j0; w1[d] = a1; w0[d] = 1.f - a1;
  }
  #pragma unroll
  for (int e = 0; e < 8; ++e) {
    int dx = (e >> 2) & 1, dy = (e >> 1) & 1, dz = e & 1;
    int g = ((j[0] + dx) * 5 + (j[1] + dy)) * 5 + (j[2] + dz);
    float w = (dx ? w1[0] : w0[0]) * (dy ? w1[1] : w0[1]) * (dz ? w1[2] : w0[2]);
    aidx[t * 8 + e] = g;
    aw[t * 8 + e] = w;
  }
}

// ---------------------------------------------------------------- p-branch -
__global__ void pconv0_kernel(const float* __restrict__ fea,
                              const float* __restrict__ Wp0,
                              float* __restrict__ y) {
  int t = blockIdx.x * blockDim.x + threadIdx.x;
  if (t >= B_ * 16 * N_) return;
  int n = t % N_;
  int d = (t / N_) % 16;
  int b = t / (16 * N_);
  float acc = 0.f;
  for (int c = 0; c < C_; ++c)
    acc += fea[(b * C_ + c) * N_ + n] * Wp0[c * 16 + d];
  y[(b * 16 + d) * N_ + n] = acc;
}

__global__ void pconv_sparse_kernel(const float* __restrict__ pin,
                                    const float* __restrict__ W,
                                    const int* __restrict__ knn,
                                    const int* __restrict__ aidx,
                                    const float* __restrict__ aw,
                                    float* __restrict__ y, int Cout) {
  int t = blockIdx.x * blockDim.x + threadIdx.x;
  if (t >= B_ * N_ * Cout) return;
  int n = t % N_;
  int d = (t / N_) % Cout;
  int b = t / (Cout * N_);
  int base = (b * N_ + n) * K_;
  float acc = 0.f;
  for (int k = 0; k < K_; ++k) {
    int nb = knn[base + k];
    float pv[16];
    #pragma unroll
    for (int c = 0; c < 16; ++c) pv[c] = pin[(b * 16 + c) * N_ + nb];
    int ab = (base + k) * 8;
    #pragma unroll
    for (int e = 0; e < 8; ++e) {
      int g = aidx[ab + e];
      float w = aw[ab + e];
      float s = 0.f;
      #pragma unroll
      for (int c = 0; c < 16; ++c) s += pv[c] * W[(g * 16 + c) * Cout + d];
      acc += w * s;
    }
  }
  y[(b * Cout + d) * N_ + n] = acc;
}

// ---------------------------------------------------------------- BN -------
__global__ void bnstats_kernel(const float* __restrict__ y, int C,
                               float* __restrict__ mean, float* __restrict__ var) {
  __shared__ float s1[256], s2[256];
  int c = blockIdx.x;
  float a = 0.f, q = 0.f;
  for (int i = threadIdx.x; i < B_ * N_; i += 256) {
    int b = i / N_, n = i - b * N_;
    float v = y[(b * C + c) * N_ + n];
    a += v; q += v * v;
  }
  s1[threadIdx.x] = a; s2[threadIdx.x] = q;
  __syncthreads();
  for (int s = 128; s > 0; s >>= 1) {
    if (threadIdx.x < s) { s1[threadIdx.x] += s1[threadIdx.x + s]; s2[threadIdx.x] += s2[threadIdx.x + s]; }
    __syncthreads();
  }
  if (threadIdx.x == 0) {
    float m = s1[0] / (float)(B_ * N_);
    mean[c] = m;
    var[c] = s2[0] / (float)(B_ * N_) - m * m;
  }
}

__global__ void bnrelu_kernel(const float* __restrict__ y,
                              const float* __restrict__ mean, const float* __restrict__ var,
                              const float* __restrict__ w, const float* __restrict__ bb,
                              float* __restrict__ out, int C) {
  int t = blockIdx.x * blockDim.x + threadIdx.x;
  if (t >= B_ * C * N_) return;
  int c = (t / N_) % C;
  float v = (y[t] - mean[c]) * rsqrtf(var[c] + EPS) * w[c] + bb[c];
  out[t] = fmaxf(v, 0.f);
}

__global__ void bnrelu_mask_kernel(const float* __restrict__ y,
                                   const float* __restrict__ mean, const float* __restrict__ var,
                                   const float* __restrict__ w, const float* __restrict__ bb,
                                   const float* __restrict__ chm, int layer,
                                   const float* __restrict__ ptm,
                                   float* __restrict__ out) {
  int t = blockIdx.x * blockDim.x + threadIdx.x;
  if (t >= B_ * C_ * N_) return;
  int n = t % N_;
  int c = (t / N_) % C_;
  int b = t / (C_ * N_);
  float v = (y[t] - mean[c]) * rsqrtf(var[c] + EPS) * w[c] + bb[c];
  v = fmaxf(v, 0.f);
  float cs = chm[(layer * C_ + c) * 2 + 1];
  float cd = chm[(layer * C_ + c) * 2 + 0];
  out[t] = v * (cs * ptm[b * N_ + n] + cd);
}

// ---------------------------------------------------------------- masks ----
__global__ void ptmask_kernel(const float* __restrict__ y2,
                              const float* __restrict__ gpt,
                              float* __restrict__ ptm) {
  int t = blockIdx.x * blockDim.x + threadIdx.x;
  if (t >= B_ * N_) return;
  int b = t / N_, n = t - b * N_;
  float a0 = y2[(b * 2 + 0) * N_ + n] + gpt[(b * 2 + 0) * N_ + n];
  float a1 = y2[(b * 2 + 1) * N_ + n] + gpt[(b * 2 + 1) * N_ + n];
  float m = fmaxf(a0, a1);
  float e0 = expf(a0 - m), e1 = expf(a1 - m);
  ptm[t] = e1 / (e0 + e1);
}

__global__ void chmask_kernel(const float* __restrict__ cmp,
                              const float* __restrict__ gch,
                              float* __restrict__ chm) {
  int t = blockIdx.x * blockDim.x + threadIdx.x;
  if (t >= 2 * C_) return;
  float a0 = cmp[t * 2 + 0] + gch[t * 2 + 0];
  float a1 = cmp[t * 2 + 1] + gch[t * 2 + 1];
  float m = fmaxf(a0, a1);
  float e0 = expf(a0 - m), e1 = expf(a1 - m);
  float s = e0 + e1;
  chm[t * 2 + 0] = e0 / s;
  chm[t * 2 + 1] = e1 / s;
}

// ------------------------------------------------------- WMMA staging ------
// (B,C,N) fp32 -> (B,N,C) fp16: neighbor gather becomes one 128B line
__global__ void cvt_half_kernel(const float* __restrict__ x, _Float16* __restrict__ xh) {
  int t = blockIdx.x * blockDim.x + threadIdx.x;
  if (t >= B_ * N_ * C_) return;
  int c = t % C_;
  int n = (t / C_) % N_;
  int b = t / (C_ * N_);
  xh[t] = (_Float16)x[(b * C_ + c) * N_ + n];
}

// W(G,64,64) fp32 -> fp16 B-fragments, LANE-MAJOR so each lane's fragment is
// 8 consecutive dwords (-> 2x global_load_b128):
//   whf[(((g*4+dt)*2+kc)*32 + lane)*8 + v] = (W[g][k0][d], W[g][k0+1][d])
//   d = dt*16 + lane%16 ; k0 = kc*32 + (lane/16)*8 + (v&3)*2 + (v>>2)*16
__global__ void prepw_kernel(const float* __restrict__ W, unsigned* __restrict__ whf) {
  int t = blockIdx.x * blockDim.x + threadIdx.x;
  if (t >= G_ * 2048) return;
  int v    = t & 7;
  int lane = (t >> 3) & 31;
  int kc   = (t >> 8) & 1;
  int dt   = (t >> 9) & 3;
  int g    = t >> 11;
  int d = dt * 16 + (lane & 15);
  int k0 = kc * 32 + ((lane >> 4) << 3) + (v & 3) * 2 + ((v >> 2) << 4);
  float lo = W[(g * 64 + k0) * 64 + d];
  float hi = W[(g * 64 + k0 + 1) * 64 + d];
  whf[t] = pk2h(lo, hi);
}

// ------------------------------------------------------- body KNN-conv -----
// 1 block = 16 points. 4 waves; wave = one 16-wide d-tile.
// Phase 1: sparse alpha scatter-accumulate into LDS acc[16][GC][64] (fp32).
// Phase 1b: pack acc into fp16 A-fragments in LDS, lane-slot layout padded to
//           12 dwords (12*m mod 64 = 4*(3m mod 16): 16 distinct bank groups,
//           16B aligned); one ds_store_b128 per 4 half2s.
// Phase 2: per WMMA: 2x ds_load_b128 (A) + 2x global_load_b128 (B) +
//          v_wmma_f32_16x16x32_f16; fp32 accum resident in VGPRs throughout.
__global__ void __launch_bounds__(128)
body_wmma_kernel(const unsigned* __restrict__ xh,   // fp16 pairs, (b,n,c)
                 const unsigned* __restrict__ whf,  // pre-swizzled fragments
                 const int* __restrict__ aidx, const float* __restrict__ aw,
                 const int* __restrict__ knn,
                 float* __restrict__ y) {           // (B,64,N) pre-BN
  extern __shared__ char smem[];
  float*    acc  = (float*)smem;                       // 16*GC*64 fp32 (32KB)
  unsigned* accH = (unsigned*)(smem + 16 * GC * 64 * 4); // GC*4*16*12 dwords (24KB)

  const int b  = blockIdx.x >> 8;                   // 256 tiles per batch
  const int n0 = (blockIdx.x & 255) << 4;
  const int tid  = threadIdx.x;
  const int wave = tid >> 5;
  const int lane = tid & 31;
  const int dt = wave;
  const int m  = lane & 15;
  const int grp = lane >> 4;

  v8f cacc = {};

  for (int g0 = 0; g0 < G_; g0 += GC) {
    int gc = G_ - g0; if (gc > GC) gc = GC;
    {
      v4u z = {};
      for (int i = tid; i < 16 * GC * 16; i += 128) ((v4u*)acc)[i] = z;
    }
    __syncthreads();

    // ---- phase 1: one wave per point (4 pts each), 2 channels per lane ----
    for (int pp = 0; pp < 4; ++pp) {
      int pt = wave * 4 + pp;
      int n = n0 + pt;
      int base = (b * N_ + n) * K_;
      for (int k = 0; k < K_; ++k) {
        int nb = knn[base + k];
        unsigned fv = xh[(b * N_ + nb) * 32 + lane];
        union { unsigned u; _Float16 h[2]; } uf; uf.u = fv;
        float f0 = (float)uf.h[0], f1 = (float)uf.h[1];
        int ab = (base + k) * 8;
        #pragma unroll
        for (int e = 0; e < 8; ++e) {
          int gl = aidx[ab + e] - g0;
          if ((unsigned)gl < (unsigned)gc) {
            float w = aw[ab + e];
            int ai = (pt * GC + gl) * 64 + 2 * lane;
            acc[ai]     += w * f0;
            acc[ai + 1] += w * f1;
          }
        }
      }
    }
    __syncthreads();

    // ---- phase 1b: fp32 acc -> fp16 A-fragments (exact WMMA lane/K order) -
    // quad (gp,kc,gr,mm,q): 4 packed dwords at accH[slot*12 + q*4], slot =
    // ((gp*2+kc)*2+gr)*16+mm ; one ds_store_b128 each.
    for (int i = tid; i < GC * 128; i += 128) {
      int qq = i & 1;
      int mm = (i >> 1) & 15;
      int gr = (i >> 5) & 1;
      int kc = (i >> 6) & 1;
      int gp = i >> 7;
      int kbase = kc * 32 + gr * 8 + qq * 16;
      int src = (mm * GC + gp) * 64 + kbase;
      v4u qv;
      #pragma unroll
      for (int v = 0; v < 4; ++v)
        qv[v] = pk2h(acc[src + v * 2], acc[src + v * 2 + 1]);
      int slot = ((gp * 2 + kc) * 2 + gr) * 16 + mm;
      *(v4u*)(accH + slot * 12 + qq * 4) = qv;
    }
    __syncthreads();

    // ---- phase 2: WMMA over the chunk's (g,c) contraction ----
    for (int gp = 0; gp < gc; ++gp) {
      int g = g0 + gp;
      if (g + 1 < G_)
        __builtin_prefetch(&whf[(((g + 1) * 4 + dt) * 2 * 32 + lane) * 8], 0, 1);
      #pragma unroll
      for (int kc = 0; kc < 2; ++kc) {
        FragU A, Bf;
        const unsigned* ap = accH + (((gp * 2 + kc) * 2 + grp) * 16 + m) * 12;
        const unsigned* bp = whf + (((g * 4 + dt) * 2 + kc) * 32 + lane) * 8;
        A.q[0]  = *(const v4u*)(ap);       // ds_load_b128
        A.q[1]  = *(const v4u*)(ap + 4);   // ds_load_b128
        Bf.q[0] = *(const v4u*)(bp);       // global_load_b128
        Bf.q[1] = *(const v4u*)(bp + 4);   // global_load_b128
        cacc = __builtin_amdgcn_wmma_f32_16x16x32_f16(
            false, A.h, false, Bf.h, (short)0, cacc, false, false);
      }
    }
    __syncthreads();
  }

  // C/D layout: VGPR r -> pt = r + (lane>=16 ? 8 : 0), d = dt*16 + lane%16
  #pragma unroll
  for (int r = 0; r < 8; ++r) {
    int pt = r + (grp << 3);
    int d = dt * 16 + m;
    y[(b * C_ + d) * N_ + n0 + pt] = cacc[r];
  }
}

// ---------------------------------------------------------------- tail -----
__global__ void tail_kernel(const float* __restrict__ o0, const float* __restrict__ o1,
                            const float* __restrict__ Wt, const float* __restrict__ bt,
                            float* __restrict__ y) {
  int t = blockIdx.x * blockDim.x + threadIdx.x;
  if (t >= B_ * C_ * N_) return;
  int n = t % N_;
  int d = (t / N_) % C_;
  int b = t / (C_ * N_);
  float acc = bt[d];
  for (int c = 0; c < C_; ++c)  acc += o0[(b * C_ + c) * N_ + n] * Wt[c * C_ + d];
  for (int c = 0; c < C_; ++c)  acc += o1[(b * C_ + c) * N_ + n] * Wt[(C_ + c) * C_ + d];
  y[t] = acc;
}

__global__ void final_kernel(const float* __restrict__ y,
                             const float* __restrict__ mean, const float* __restrict__ var,
                             const float* __restrict__ w, const float* __restrict__ bb,
                             const float* __restrict__ fea, float* __restrict__ out) {
  int t = blockIdx.x * blockDim.x + threadIdx.x;
  if (t >= B_ * C_ * N_) return;
  int c = (t / N_) % C_;
  float v = (y[t] - mean[c]) * rsqrtf(var[c] + EPS) * w[c] + bb[c];
  out[t] = fmaxf(v + fea[t], 0.f);
}

// ---------------------------------------------------------------- flops ----
__global__ void flops_kernel(const float* __restrict__ ptm, const float* __restrict__ chm,
                             float* __restrict__ out) {
  int t = blockIdx.x * blockDim.x + threadIdx.x;
  const int per = B_ * C_ * N_;
  if (t >= 2 * per) return;
  int l = t / per;
  int r = t - l * per;
  int n = r % N_;
  int c = (r / N_) % C_;
  int b = r / (C_ * N_);
  float cs = chm[(l * C_ + c) * 2 + 1];
  float cd = chm[(l * C_ + c) * 2 + 0];
  out[per + t] = (ptm[b * N_ + n] * cs + cd) * (float)(K_ * (C_ + 1)); // 1040
  if (t == 0) out[per + 2 * per] = 1090519040.0f; // 2 * B*N*K*C*(C+1), exact in fp32
}

// ================================================================ host =====
extern "C" void kernel_launch(void* const* d_in, const int* in_sizes, int n_in,
                              void* d_out, int out_size, void* d_ws, size_t ws_size,
                              hipStream_t stream) {
  const float* xyz  = (const float*)d_in[0];
  const float* fea  = (const float*)d_in[1];
  const int*   knn  = (const int*)  d_in[2];
  const float* Wb0  = (const float*)d_in[3];
  const float* bnw0 = (const float*)d_in[4];
  const float* bnb0 = (const float*)d_in[5];
  const float* Wb1  = (const float*)d_in[6];
  const float* bnw1 = (const float*)d_in[7];
  const float* bnb1 = (const float*)d_in[8];
  const float* Wp0  = (const float*)d_in[9];
  const float* bnpw0= (const float*)d_in[10];
  const float* bnpb0= (const float*)d_in[11];
  const float* Wp1  = (const float*)d_in[12];
  const float* bnpw1= (const float*)d_in[13];
  const float* bnpb1= (const float*)d_in[14];
  const float* Wp2  = (const float*)d_in[15];
  const float* cmp  = (const float*)d_in[16];
  const float* Wt   = (const float*)d_in[17];
  const float* bt   = (const float*)d_in[18];
  const float* bntw = (const float*)d_in[19];
  const float* bntb = (const float*)d_in[20];
  const float* gpt  = (const float*)d_in[21];
  const float* gch  = (const float*)d_in[22];
  float* out = (float*)d_out;

  char* p = (char*)d_ws;
  auto alloc = [&](size_t n) { char* r = p; p += (n + 255) & ~(size_t)255; return (void*)r; };
  float*    mean = (float*)alloc(64 * 4);
  float*    var  = (float*)alloc(64 * 4);
  float*    chm  = (float*)alloc(2 * C_ * 2 * 4);
  float*    ptm  = (float*)alloc((size_t)B_ * N_ * 4);
  float*    P0   = (float*)alloc((size_t)B_ * 16 * N_ * 4);
  float*    P1   = (float*)alloc((size_t)B_ * 16 * N_ * 4);
  float*    ybuf = (float*)alloc((size_t)B_ * C_ * N_ * 4);
  float*    O0   = (float*)alloc((size_t)B_ * C_ * N_ * 4);
  float*    O1   = (float*)alloc((size_t)B_ * C_ * N_ * 4);
  int*      aidx = (int*)  alloc((size_t)B_ * N_ * K_ * 8 * 4);
  float*    aw   = (float*)alloc((size_t)B_ * N_ * K_ * 8 * 4);
  _Float16* xh   = (_Float16*)alloc((size_t)B_ * N_ * C_ * 2);
  unsigned* whf  = (unsigned*)alloc((size_t)G_ * 2048 * 4);

  const int T = 256;
  auto blk = [](int n, int t) { return (n + t - 1) / t; };

  // sparse trilinear alpha + masks
  alpha_kernel<<<blk(B_ * N_ * K_, T), T, 0, stream>>>(xyz, knn, aidx, aw);
  chmask_kernel<<<1, 128, 0, stream>>>(cmp, gch, chm);

  // p-branch (C=16, VALU)
  pconv0_kernel<<<blk(B_ * 16 * N_, T), T, 0, stream>>>(fea, Wp0, ybuf);
  bnstats_kernel<<<16, 256, 0, stream>>>(ybuf, 16, mean, var);
  bnrelu_kernel<<<blk(B_ * 16 * N_, T), T, 0, stream>>>(ybuf, mean, var, bnpw0, bnpb0, P0, 16);
  pconv_sparse_kernel<<<blk(B_ * N_ * 16, T), T, 0, stream>>>(P0, Wp1, knn, aidx, aw, ybuf, 16);
  bnstats_kernel<<<16, 256, 0, stream>>>(ybuf, 16, mean, var);
  bnrelu_kernel<<<blk(B_ * 16 * N_, T), T, 0, stream>>>(ybuf, mean, var, bnpw1, bnpb1, P1, 16);
  pconv_sparse_kernel<<<blk(B_ * N_ * 2, T), T, 0, stream>>>(P1, Wp2, knn, aidx, aw, ybuf, 2);
  ptmask_kernel<<<blk(B_ * N_, T), T, 0, stream>>>(ybuf, gpt, ptm);

  const int nTiles = B_ * (N_ / 16);                        // 512
  const size_t smem = (size_t)16 * GC * 64 * 4              // fp32 acc
                    + (size_t)GC * 4 * 16 * 12 * 4;         // fp16 A-frags (56KB)

  // body layer 0 (WMMA)
  cvt_half_kernel<<<blk(B_ * N_ * C_, T), T, 0, stream>>>(fea, xh);
  prepw_kernel<<<blk(G_ * 2048, T), T, 0, stream>>>(Wb0, whf);
  body_wmma_kernel<<<nTiles, 128, smem, stream>>>((const unsigned*)xh, whf, aidx, aw, knn, ybuf);
  bnstats_kernel<<<64, 256, 0, stream>>>(ybuf, 64, mean, var);
  bnrelu_mask_kernel<<<blk(B_ * C_ * N_, T), T, 0, stream>>>(ybuf, mean, var, bnw0, bnb0, chm, 0, ptm, O0);

  // body layer 1 (WMMA)
  cvt_half_kernel<<<blk(B_ * N_ * C_, T), T, 0, stream>>>(O0, xh);
  prepw_kernel<<<blk(G_ * 2048, T), T, 0, stream>>>(Wb1, whf);
  body_wmma_kernel<<<nTiles, 128, smem, stream>>>((const unsigned*)xh, whf, aidx, aw, knn, ybuf);
  bnstats_kernel<<<64, 256, 0, stream>>>(ybuf, 64, mean, var);
  bnrelu_mask_kernel<<<blk(B_ * C_ * N_, T), T, 0, stream>>>(ybuf, mean, var, bnw1, bnb1, chm, 1, ptm, O1);

  // tail + residual
  tail_kernel<<<blk(B_ * C_ * N_, T), T, 0, stream>>>(O0, O1, Wt, bt, ybuf);
  bnstats_kernel<<<64, 256, 0, stream>>>(ybuf, 64, mean, var);
  final_kernel<<<blk(B_ * C_ * N_, T), T, 0, stream>>>(ybuf, mean, var, bntw, bntb, fea, out);

  // flops outputs + total scalar
  flops_kernel<<<blk(2 * B_ * C_ * N_, T), T, 0, stream>>>(ptm, chm, out);
}
